// Attention_20323785244740
// MI455X (gfx1250) — matrix operand order repaired and checked by reading, not compiled, not run on hardware
//
#include <hip/hip_runtime.h>
#include <hip/hip_bf16.h>

// ---------------------------------------------------------------------------
// ViT attention block on MI455X (gfx1250, wave32, WMMA).
//   out = proj( softmax(QK^T * scale) V ) + b,  QKV = x @ W_qkv^T
// All matmuls run through v_wmma_f32_16x16x32_f16 (f32 accumulate).
// GEMM waves own 32x64 output tiles: 8 WMMA per 12 b128 loads per k-step.
// ---------------------------------------------------------------------------

#define BATCH     32
#define SEQ       577
#define DIM       768
#define NUM_HEADS 12
#define HEAD      64
#define NP        608          // SEQ padded up to a multiple of 32
#define MTOT      (BATCH * SEQ) // 18464 token rows
#define SCALE     0.125f       // 64^-0.5

typedef __attribute__((ext_vector_type(8)))  _Float16 v8h;
typedef __attribute__((ext_vector_type(16))) _Float16 v16h;
typedef __attribute__((ext_vector_type(8)))  float    v8f;

#define DEVINL __device__ __forceinline__

DEVINL v16h cat8(v8h lo, v8h hi) {
  return __builtin_shufflevector(lo, hi, 0,1,2,3,4,5,6,7,8,9,10,11,12,13,14,15);
}

// A fragment (16x32 f16, M x K). Caller passes this lane's row pointer advanced
// to the K-chunk. Lane l (0-15): K = {0..7,16..23}; lane l+16: K = {8..15,24..31}.
DEVINL v16h load_a_frag(const _Float16* rowk, int lane) {
  const int o = (lane >> 4) * 8;
  v8h lo = *(const v8h*)(rowk + o);
  v8h hi = *(const v8h*)(rowk + o + 16);
  return cat8(lo, hi);
}

// B fragment (32x16 f16, K x N). Lane l (0-15): K = {0..7,16..23}; lane l+16:
// K = {8..15,24..31} (same order as A).
DEVINL v16h load_b_frag(const _Float16* rowk, int lane) {
  const int o = (lane >> 4) * 8;
  v8h lo = *(const v8h*)(rowk + o);
  v8h hi = *(const v8h*)(rowk + o + 16);
  return cat8(lo, hi);
}

DEVINL v8f wmma_f16(v16h a, v16h b, v8f c) {
  return __builtin_amdgcn_wmma_f32_16x16x32_f16(false, a, false, b, (short)0, c,
                                                false, false);
}

// ---------------------------------------------------------------------------
// fp32 -> fp16 cast (grid-stride)
// ---------------------------------------------------------------------------
__global__ void cast_f32_to_f16(const float* __restrict__ in,
                                _Float16* __restrict__ out, int n) {
  int i = blockIdx.x * blockDim.x + threadIdx.x;
  const int stride = gridDim.x * blockDim.x;
  for (; i < n; i += stride) out[i] = (_Float16)in[i];
}

// ---------------------------------------------------------------------------
// QKV GEMM: [18464 x 2304] = Xh[18464 x 768] * Wqkv[2304 x 768]^T,
// scattered into Q[B,H,NP,D], K[B,H,NP,D], Vt[B,H,D,NP]  (all f16).
// Block = 4 waves stacked along M; wave tile = 32 rows x 64 cols.
// ---------------------------------------------------------------------------
__global__ __launch_bounds__(128)
void qkv_gemm_kernel(const _Float16* __restrict__ X,
                     const _Float16* __restrict__ W,
                     _Float16* __restrict__ Q,
                     _Float16* __restrict__ Kt,
                     _Float16* __restrict__ Vt) {
  const int lane = threadIdx.x;
  const int wave = threadIdx.y;
  const int n0 = blockIdx.x * 64;                 // 64 output cols per wave
  const int mw = blockIdx.y * 128 + wave * 32;    // 32 output rows per wave

  // Clamp A-row loads in-bounds (stores are guarded below).
  int r0 = mw + (lane & 15);       r0 = r0 < MTOT ? r0 : MTOT - 1;
  int r1 = mw + 16 + (lane & 15);  r1 = r1 < MTOT ? r1 : MTOT - 1;
  const _Float16* a0 = X + (size_t)r0 * DIM;
  const _Float16* a1 = X + (size_t)r1 * DIM;
  const _Float16* b0 = W + (size_t)(n0 +  0 + (lane & 15)) * DIM;
  const _Float16* b1 = W + (size_t)(n0 + 16 + (lane & 15)) * DIM;
  const _Float16* b2 = W + (size_t)(n0 + 32 + (lane & 15)) * DIM;
  const _Float16* b3 = W + (size_t)(n0 + 48 + (lane & 15)) * DIM;

  v8f acc[2][4] = {};
#pragma unroll 2
  for (int k0 = 0; k0 < DIM; k0 += 32) {
    __builtin_prefetch(a0 + k0 + 128, 0, 1);
    const v16h af0 = load_a_frag(a0 + k0, lane);
    const v16h af1 = load_a_frag(a1 + k0, lane);
    const v16h bf0 = load_b_frag(b0 + k0, lane);
    const v16h bf1 = load_b_frag(b1 + k0, lane);
    const v16h bf2 = load_b_frag(b2 + k0, lane);
    const v16h bf3 = load_b_frag(b3 + k0, lane);
    acc[0][0] = wmma_f16(af0, bf0, acc[0][0]);
    acc[0][1] = wmma_f16(af0, bf1, acc[0][1]);
    acc[0][2] = wmma_f16(af0, bf2, acc[0][2]);
    acc[0][3] = wmma_f16(af0, bf3, acc[0][3]);
    acc[1][0] = wmma_f16(af1, bf0, acc[1][0]);
    acc[1][1] = wmma_f16(af1, bf1, acc[1][1]);
    acc[1][2] = wmma_f16(af1, bf2, acc[1][2]);
    acc[1][3] = wmma_f16(af1, bf3, acc[1][3]);
  }

  // Column decomposition (per j sub-tile): which = c/768, h = (c%768)/64, d = c%64.
  int whichJ[4], hJ[4], dJ[4];
#pragma unroll
  for (int j = 0; j < 4; ++j) {
    const int c = n0 + j * 16 + (lane & 15);
    whichJ[j] = c / DIM;
    hJ[j]     = (c % DIM) / HEAD;
    dJ[j]     = c % HEAD;
  }

#pragma unroll
  for (int i = 0; i < 2; ++i) {
#pragma unroll
    for (int r = 0; r < 8; ++r) {
      const int m = mw + i * 16 + r + ((lane >> 4) << 3);
      if (m >= MTOT) continue;
      const int bidx = m / SEQ;           // batch
      const int n    = m % SEQ;           // token
#pragma unroll
      for (int j = 0; j < 4; ++j) {
        const size_t bh = (size_t)(bidx * NUM_HEADS + hJ[j]);
        const _Float16 v = (_Float16)acc[i][j][r];
        if (whichJ[j] == 0)      Q [(bh * NP + n) * HEAD + dJ[j]] = v;
        else if (whichJ[j] == 1) Kt[(bh * NP + n) * HEAD + dJ[j]] = v;
        else                     Vt[(bh * HEAD + dJ[j]) * NP + n] = v;
      }
    }
  }
}

// ---------------------------------------------------------------------------
// Flash-style attention. One wave owns a 16-query tile; streams keys 32/iter.
//   S = Q K^T * scale (4 WMMAs/iter), online softmax, O += P V (4 WMMAs/iter).
// P is bounced through LDS to re-pack into A-fragment layout.
// ---------------------------------------------------------------------------
__global__ __launch_bounds__(128)
void attn_kernel(const _Float16* __restrict__ Q,
                 const _Float16* __restrict__ K,
                 const _Float16* __restrict__ Vt,
                 _Float16* __restrict__ Ao) {
  __shared__ _Float16 pbuf[4][16 * 32];

  const int lane = threadIdx.x;
  const int wave = threadIdx.y;
  const int bh   = blockIdx.x;                 // batch*head
  const int t    = blockIdx.y * 4 + wave;      // query tile index
  if (t >= (SEQ + 15) / 16) return;

  const int b  = bh / NUM_HEADS;
  const int h  = bh % NUM_HEADS;
  const int m0 = t * 16;
  const int colA = lane & 15;

  const _Float16* Qb = Q  + (size_t)bh * NP * HEAD;
  const _Float16* Kb = K  + (size_t)bh * NP * HEAD;
  const _Float16* Vb = Vt + (size_t)bh * HEAD * NP;

  // Q fragments for the whole D=64 (two 16x32 A-frags), reused every iter.
  const _Float16* qrow = Qb + (size_t)(m0 + colA) * HEAD;
  const v16h qf0 = load_a_frag(qrow + 0,  lane);
  const v16h qf1 = load_a_frag(qrow + 32, lane);

  v8f o0 = {}, o1 = {}, o2 = {}, o3 = {};
  float mrow[8], lrow[8];
#pragma unroll
  for (int r = 0; r < 8; ++r) { mrow[r] = -1e30f; lrow[r] = 0.0f; }

  _Float16* pl = pbuf[wave];

  for (int j0 = 0; j0 < NP; j0 += 32) {
    // ---- scores: S0 = keys j0..j0+15, S1 = keys j0+16..j0+31 -------------
    const _Float16* k0row = Kb + (size_t)(j0 + colA) * HEAD;
    const _Float16* k1row = Kb + (size_t)(j0 + 16 + colA) * HEAD;
    v8f s0 = {}, s1 = {};
    s0 = wmma_f16(qf0, load_b_frag(k0row + 0,  lane), s0);
    s0 = wmma_f16(qf1, load_b_frag(k0row + 32, lane), s0);
    s1 = wmma_f16(qf0, load_b_frag(k1row + 0,  lane), s1);
    s1 = wmma_f16(qf1, load_b_frag(k1row + 32, lane), s1);

    const bool ok0 = (j0 + colA)      < SEQ;
    const bool ok1 = (j0 + 16 + colA) < SEQ;

    // ---- online softmax (row reductions across 16-lane column groups) ----
#pragma unroll
    for (int r = 0; r < 8; ++r) {
      float a0 = ok0 ? s0[r] * SCALE : -1e30f;
      float a1 = ok1 ? s1[r] * SCALE : -1e30f;
      float sm = fmaxf(a0, a1);
      sm = fmaxf(sm, __shfl_xor(sm, 1, 32));
      sm = fmaxf(sm, __shfl_xor(sm, 2, 32));
      sm = fmaxf(sm, __shfl_xor(sm, 4, 32));
      sm = fmaxf(sm, __shfl_xor(sm, 8, 32));
      const float mnew = fmaxf(mrow[r], sm);
      const float corr = __expf(mrow[r] - mnew);
      const float p0 = __expf(a0 - mnew);
      const float p1 = __expf(a1 - mnew);
      float rs = p0 + p1;
      rs += __shfl_xor(rs, 1, 32);
      rs += __shfl_xor(rs, 2, 32);
      rs += __shfl_xor(rs, 4, 32);
      rs += __shfl_xor(rs, 8, 32);
      lrow[r] = lrow[r] * corr + rs;
      mrow[r] = mnew;
      o0[r] *= corr; o1[r] *= corr; o2[r] *= corr; o3[r] *= corr;

      const int prow = r + ((lane >> 4) << 3);       // row this lane holds
      pl[prow * 32 + colA]      = (_Float16)p0;
      pl[prow * 32 + 16 + colA] = (_Float16)p1;
    }

    // ---- re-pack P (16x32) from LDS into A-fragment layout ---------------
    const _Float16* prow_ptr = pl + (size_t)colA * 32;
    const int o  = (lane >> 4) * 8;
    v8h plo = *(const v8h*)(prow_ptr + o);
    v8h phi = *(const v8h*)(prow_ptr + o + 16);
    const v16h pf = cat8(plo, phi);

    // ---- O += P * V  (V pre-transposed: contiguous along keys) -----------
    o0 = wmma_f16(pf, load_b_frag(Vb + (size_t)( 0 + colA) * NP + j0, lane), o0);
    o1 = wmma_f16(pf, load_b_frag(Vb + (size_t)(16 + colA) * NP + j0, lane), o1);
    o2 = wmma_f16(pf, load_b_frag(Vb + (size_t)(32 + colA) * NP + j0, lane), o2);
    o3 = wmma_f16(pf, load_b_frag(Vb + (size_t)(48 + colA) * NP + j0, lane), o3);
  }

  // ---- epilogue: normalize, write token-major f16 [B,N,C] ----------------
#pragma unroll
  for (int r = 0; r < 8; ++r) {
    const int m = m0 + r + ((lane >> 4) << 3);
    if (m < SEQ) {
      const float inv = lrow[r] > 0.0f ? 1.0f / lrow[r] : 0.0f;
      const size_t base = ((size_t)(b * SEQ + m)) * DIM + h * HEAD;
      Ao[base +  0 + colA] = (_Float16)(o0[r] * inv);
      Ao[base + 16 + colA] = (_Float16)(o1[r] * inv);
      Ao[base + 32 + colA] = (_Float16)(o2[r] * inv);
      Ao[base + 48 + colA] = (_Float16)(o3[r] * inv);
    }
  }
}

// ---------------------------------------------------------------------------
// Output projection: out[18464 x 768] (f32) = A[18464 x 768] * Wp[768x768]^T + b
// Same 32x64 wave tiling as the QKV GEMM.
// ---------------------------------------------------------------------------
__global__ __launch_bounds__(128)
void proj_gemm_kernel(const _Float16* __restrict__ A,
                      const _Float16* __restrict__ W,
                      const float* __restrict__ bias,
                      float* __restrict__ out) {
  const int lane = threadIdx.x;
  const int wave = threadIdx.y;
  const int n0 = blockIdx.x * 64;
  const int mw = blockIdx.y * 128 + wave * 32;

  int r0 = mw + (lane & 15);       r0 = r0 < MTOT ? r0 : MTOT - 1;
  int r1 = mw + 16 + (lane & 15);  r1 = r1 < MTOT ? r1 : MTOT - 1;
  const _Float16* a0 = A + (size_t)r0 * DIM;
  const _Float16* a1 = A + (size_t)r1 * DIM;
  const _Float16* b0 = W + (size_t)(n0 +  0 + (lane & 15)) * DIM;
  const _Float16* b1 = W + (size_t)(n0 + 16 + (lane & 15)) * DIM;
  const _Float16* b2 = W + (size_t)(n0 + 32 + (lane & 15)) * DIM;
  const _Float16* b3 = W + (size_t)(n0 + 48 + (lane & 15)) * DIM;

  v8f acc[2][4] = {};
#pragma unroll 2
  for (int k0 = 0; k0 < DIM; k0 += 32) {
    __builtin_prefetch(a0 + k0 + 128, 0, 1);
    const v16h af0 = load_a_frag(a0 + k0, lane);
    const v16h af1 = load_a_frag(a1 + k0, lane);
    const v16h bf0 = load_b_frag(b0 + k0, lane);
    const v16h bf1 = load_b_frag(b1 + k0, lane);
    const v16h bf2 = load_b_frag(b2 + k0, lane);
    const v16h bf3 = load_b_frag(b3 + k0, lane);
    acc[0][0] = wmma_f16(af0, bf0, acc[0][0]);
    acc[0][1] = wmma_f16(af0, bf1, acc[0][1]);
    acc[0][2] = wmma_f16(af0, bf2, acc[0][2]);
    acc[0][3] = wmma_f16(af0, bf3, acc[0][3]);
    acc[1][0] = wmma_f16(af1, bf0, acc[1][0]);
    acc[1][1] = wmma_f16(af1, bf1, acc[1][1]);
    acc[1][2] = wmma_f16(af1, bf2, acc[1][2]);
    acc[1][3] = wmma_f16(af1, bf3, acc[1][3]);
  }

#pragma unroll
  for (int i = 0; i < 2; ++i) {
#pragma unroll
    for (int r = 0; r < 8; ++r) {
      const int m = mw + i * 16 + r + ((lane >> 4) << 3);
      if (m >= MTOT) continue;
#pragma unroll
      for (int j = 0; j < 4; ++j) {
        const int col = n0 + j * 16 + (lane & 15);
        out[(size_t)m * DIM + col] = acc[i][j][r] + bias[col];
      }
    }
  }
}

// ---------------------------------------------------------------------------
// Launch
// ---------------------------------------------------------------------------
extern "C" void kernel_launch(void* const* d_in, const int* in_sizes, int n_in,
                              void* d_out, int out_size, void* d_ws, size_t ws_size,
                              hipStream_t stream) {
  (void)in_sizes; (void)n_in; (void)out_size;

  const float* x      = (const float*)d_in[0];   // [32,577,768]
  const float* W_qkv  = (const float*)d_in[1];   // [2304,768]
  const float* W_proj = (const float*)d_in[2];   // [768,768]
  const float* b_proj = (const float*)d_in[3];   // [768]
  float* out = (float*)d_out;

  const size_t XH_ELEMS    = (size_t)MTOT * DIM;
  const size_t WQKV_ELEMS  = (size_t)3 * DIM * DIM;
  const size_t WPROJ_ELEMS = (size_t)DIM * DIM;
  const size_t QKV_ELEMS   = (size_t)BATCH * NUM_HEADS * NP * HEAD;    // per tensor
  const size_t AT_ELEMS    = (size_t)MTOT * DIM;

  char* p = (char*)d_ws;
  auto take = [&](size_t elems) -> _Float16* {
    _Float16* r = (_Float16*)p;
    size_t bytes = (elems * sizeof(_Float16) + 255) & ~(size_t)255;
    p += bytes;
    return r;
  };
  _Float16* xh     = take(XH_ELEMS);
  _Float16* wqkvh  = take(WQKV_ELEMS);
  _Float16* wprojh = take(WPROJ_ELEMS);
  _Float16* Qf     = take(QKV_ELEMS);
  _Float16* Kf     = take(QKV_ELEMS);
  _Float16* Vt     = take(QKV_ELEMS);
  _Float16* attnh  = take(AT_ELEMS);
  if ((size_t)(p - (char*)d_ws) > ws_size) return;  // workspace too small

  // Zero-fill Q/K/Vt so the NP padding rows/cols are exact zeros.
  hipMemsetAsync(Qf, 0, QKV_ELEMS * sizeof(_Float16), stream);
  hipMemsetAsync(Kf, 0, QKV_ELEMS * sizeof(_Float16), stream);
  hipMemsetAsync(Vt, 0, QKV_ELEMS * sizeof(_Float16), stream);

  // fp32 -> fp16 casts
  cast_f32_to_f16<<<1024, 256, 0, stream>>>(x,      xh,     (int)XH_ELEMS);
  cast_f32_to_f16<<<512,  256, 0, stream>>>(W_qkv,  wqkvh,  (int)WQKV_ELEMS);
  cast_f32_to_f16<<<256,  256, 0, stream>>>(W_proj, wprojh, (int)WPROJ_ELEMS);

  // QKV projection (18464 x 2304 x 768): blocks of 128 rows x 64 cols.
  qkv_gemm_kernel<<<dim3(3 * DIM / 64, (MTOT + 127) / 128), dim3(32, 4), 0, stream>>>(
      xh, wqkvh, Qf, Kf, Vt);

  // Attention: 384 (batch*head) x 37 query tiles (4 tiles / block)
  attn_kernel<<<dim3(BATCH * NUM_HEADS, 10), dim3(32, 4), 0, stream>>>(
      Qf, Kf, Vt, attnh);

  // Output projection + bias (18464 x 768 x 768), f32 out
  proj_gemm_kernel<<<dim3(DIM / 64, (MTOT + 127) / 128), dim3(32, 4), 0, stream>>>(
      attnh, wprojh, b_proj, out);
}
